// Net_33569464385811
// MI455X (gfx1250) — 8-bit matrix operand order repaired and checked by reading, compiled, not run on hardware
//
#include <hip/hip_runtime.h>
#include <stdint.h>

typedef int   v16i __attribute__((ext_vector_type(16)));
typedef float v8f  __attribute__((ext_vector_type(8)));

#define BATCH 8192
#define IN_F  16384
#define HID   384
#define OUT_N 10
#define BK    128
#define LDSTRIDE (BK + 16)

__global__ __launch_bounds__(256) void binarize_fp8(const float* __restrict__ in,
                                                    unsigned char* __restrict__ out,
                                                    int n4) {
  int i = blockIdx.x * blockDim.x + threadIdx.x;
  int stride = gridDim.x * blockDim.x;
  const float4* in4 = (const float4*)in;
  uchar4* out4 = (uchar4*)out;
  for (; i < n4; i += stride) {
    float4 v = in4[i];
    uchar4 o;
    o.x = v.x >= 0.f ? 0x38 : 0xB8;
    o.y = v.y >= 0.f ? 0x38 : 0xB8;
    o.z = v.z >= 0.f ? 0x38 : 0xB8;
    o.w = v.w >= 0.f ? 0x38 : 0xB8;
    out4[i] = o;
  }
}

__global__ __launch_bounds__(256) void gemm1_fp8_wmma(const unsigned char* __restrict__ xb,
                                                      const unsigned char* __restrict__ w1b,
                                                      float* __restrict__ h) {
  __shared__ __align__(16) unsigned char ldsB[HID * LDSTRIDE];

  const int t    = threadIdx.x;
  const int lane = t & 31;
  const int wave = t >> 5;
  const int mIdx = wave & 3;
  const int nIdx = wave >> 2;
  const int m0   = blockIdx.x * 64 + mIdx * 16;
  const int nn   = lane & 15;
  const int hf   = lane >> 4;

  v8f acc[12] = {};

  const unsigned long long* aRow =
      (const unsigned long long*)(xb + (size_t)(m0 + nn) * IN_F);

  const unsigned ldsBase = (unsigned)(uintptr_t)(&ldsB[0]);

  for (int kb = 0; kb < IN_F; kb += BK) {
#pragma unroll
    for (int i = 0; i < 12; ++i) {
      int id = i * 256 + t;
      int n  = id >> 3;
      int c  = (id & 7) * 16;
      unsigned lds = ldsBase + (unsigned)(n * LDSTRIDE + c);
      unsigned long long ga =
          (unsigned long long)(uintptr_t)(w1b + (size_t)n * IN_F + kb + c);
      asm volatile("global_load_async_to_lds_b128 %0, %1, off"
                   :: "v"(lds), "v"(ga) : "memory");
    }
    asm volatile("s_wait_asynccnt 0" ::: "memory");
    __syncthreads();

    union { v16i v; unsigned long long d[8]; } af;
#pragma unroll
    for (int p = 0; p < 8; ++p) {
      int off = kb + (p >> 1) * 32 + hf * 16 + (p & 1) * 8;
      af.d[p] = aRow[off >> 3];
    }

#pragma unroll
    for (int g = 0; g < 12; ++g) {
      int n = (nIdx * 12 + g) * 16 + nn;
      union { v16i v; int4 q[4]; } bf;
#pragma unroll
      for (int q = 0; q < 4; ++q)
        bf.q[q] = *(const int4*)&ldsB[n * LDSTRIDE + q * 32 + hf * 16];
      acc[g] = __builtin_amdgcn_wmma_f32_16x16x128_fp8_fp8(
          af.v, bf.v, (short)0, acc[g], false, false);
    }
    __syncthreads();
  }

#pragma unroll
  for (int g = 0; g < 12; ++g) {
    int ncol = (nIdx * 12 + g) * 16 + nn;
#pragma unroll
    for (int r = 0; r < 8; ++r) {
      int row = m0 + hf * 8 + r;
      h[(size_t)row * HID + ncol] = acc[g][r];
    }
  }
}

__global__ __launch_bounds__(256) void colstats(const float* __restrict__ h,
                                                float* __restrict__ mean,
                                                float* __restrict__ var) {
  __shared__ float s1[256];
  __shared__ float s2[256];
  const int j = blockIdx.x;
  float a1 = 0.f, a2 = 0.f;
  for (int i = threadIdx.x; i < BATCH; i += 256) {
    float v = h[(size_t)i * HID + j];
    a1 += v;
    a2 += v * v;
  }
  s1[threadIdx.x] = a1;
  s2[threadIdx.x] = a2;
  __syncthreads();
  for (int s = 128; s > 0; s >>= 1) {
    if (threadIdx.x < s) {
      s1[threadIdx.x] += s1[threadIdx.x + s];
      s2[threadIdx.x] += s2[threadIdx.x + s];
    }
    __syncthreads();
  }
  if (threadIdx.x == 0) {
    float m = s1[0] * (1.f / BATCH);
    mean[j] = m;
    var[j] = s2[0] * (1.f / BATCH) - m * m;
  }
}

__global__ __launch_bounds__(256) void head10(const float* __restrict__ h,
                                              const float* __restrict__ mean,
                                              const float* __restrict__ var,
                                              const float* __restrict__ gamma,
                                              const float* __restrict__ beta,
                                              const float* __restrict__ W4,
                                              float* __restrict__ out) {
  __shared__ float aS[HID];
  __shared__ float bS[HID];
  __shared__ signed char wS[OUT_N * HID];
  for (int j = threadIdx.x; j < HID; j += 256) {
    float rstd = rsqrtf(var[j] + 1e-5f);
    float a = gamma[j] * rstd;
    aS[j] = a;
    bS[j] = beta[j] - mean[j] * a;
  }
  for (int j = threadIdx.x; j < OUT_N * HID; j += 256)
    wS[j] = W4[j] >= 0.f ? 1 : -1;
  __syncthreads();

  const int row = blockIdx.x * 256 + threadIdx.x;
  const float* hr = h + (size_t)row * HID;
  int acc[OUT_N] = {};
  for (int j = 0; j < HID; ++j) {
    int s = (aS[j] * hr[j] + bS[j]) >= 0.f ? 1 : -1;
#pragma unroll
    for (int o = 0; o < OUT_N; ++o) acc[o] += s * (int)wS[o * HID + j];
  }
#pragma unroll
  for (int o = 0; o < OUT_N; ++o)
    out[(size_t)row * OUT_N + o] = (float)acc[o];
}

extern "C" void kernel_launch(void* const* d_in, const int* in_sizes, int n_in,
                              void* d_out, int out_size, void* d_ws, size_t ws_size,
                              hipStream_t stream) {
  (void)in_sizes; (void)n_in; (void)out_size; (void)ws_size;
  const float* x     = (const float*)d_in[0];
  const float* W1    = (const float*)d_in[1];
  const float* gamma = (const float*)d_in[2];
  const float* beta  = (const float*)d_in[3];
  const float* W4    = (const float*)d_in[4];
  float* out = (float*)d_out;

  unsigned char* ws   = (unsigned char*)d_ws;
  unsigned char* xbuf = ws;
  unsigned char* w1b  = xbuf + (size_t)BATCH * IN_F;
  float*         h    = (float*)(w1b + (size_t)HID * IN_F);
  float*         mn   = h + (size_t)BATCH * HID;
  float*         vr   = mn + HID;

  binarize_fp8<<<2048, 256, 0, stream>>>(x, xbuf, BATCH * IN_F / 4);
  binarize_fp8<<<512, 256, 0, stream>>>(W1, w1b, HID * IN_F / 4);
  gemm1_fp8_wmma<<<BATCH / 64, 256, 0, stream>>>(xbuf, w1b, h);
  colstats<<<HID, 256, 0, stream>>>(h, mn, vr);
  head10<<<BATCH / 256, 256, 0, stream>>>(h, mn, vr, gamma, beta, W4, out);
}
